// BertSelfAttention_79577154060613
// MI455X (gfx1250) — matrix operand order repaired and checked by reading, not compiled, not run on hardware
//
#include <hip/hip_runtime.h>

#define B_    16
#define S_    768
#define H_    768
#define NH_   12
#define DH_   64
#define T_    128
#define QLEN_ 640   // CDD * L
#define NKT_  (S_ / 32)   // 24 key tiles

typedef __attribute__((ext_vector_type(16))) __bf16 v16bf;
typedef __attribute__((ext_vector_type(8)))  float  v8f;

typedef unsigned int tdm_v4u __attribute__((ext_vector_type(4)));
typedef int          tdm_v8i __attribute__((ext_vector_type(8)));
typedef int          tdm_v4i __attribute__((ext_vector_type(4)));

union Frag { unsigned int u[8]; v16bf v; };

__device__ __forceinline__ unsigned short f2bf(float f) {
    unsigned int u = __builtin_bit_cast(unsigned int, f);
    u += 0x7FFFu + ((u >> 16) & 1u);          // round-to-nearest-even
    return (unsigned short)(u >> 16);
}
__device__ __forceinline__ unsigned int packbf2(float a, float b) {
    return (unsigned int)f2bf(a) | ((unsigned int)f2bf(b) << 16);
}

#if __has_builtin(__builtin_amdgcn_tensor_load_to_lds)
#define HAVE_TDM 1
// Constant part of the D# (group 1): tile dims / strides (folds to immediates).
__device__ __forceinline__ tdm_v8i tdm_g1(unsigned int tile_w, unsigned int tile_h,
                                          unsigned int row_stride)
{
    const unsigned int dim_big = 1u << 20;                   // tensor dims: huge (no OOB)
    tdm_v8i g1;
    g1[0] = 0x00010000;                                      // wg_mask=0, data_size=1 (2B)
    g1[1] = (int)((dim_big & 0xFFFFu) << 16);                // tensor_dim0[15:0]
    g1[2] = (int)(((dim_big >> 16) & 0xFFFFu)                // tensor_dim0[31:16]
          | ((dim_big & 0xFFFFu) << 16));                    // tensor_dim1[15:0]
    g1[3] = (int)(((dim_big >> 16) & 0xFFFFu)                // tensor_dim1[31:16]
          | (tile_w << 16));                                 // tile_dim0
    g1[4] = (int)tile_h;                                     // tile_dim1 (tile_dim2=0)
    g1[5] = (int)row_stride;                                 // tensor_dim0_stride[31:0]
    g1[6] = 0;
    g1[7] = 0;
    return g1;
}
// Issue one async 2D tile load (group 0 rebuilt per call: lds + global addr).
__device__ __forceinline__ void tdm_issue(unsigned int lds_off, const void* gaddr,
                                          tdm_v8i g1)
{
    const unsigned long long ga = (unsigned long long)gaddr;
    tdm_v4u g0;
    g0[0] = 1u;                                              // count=1, user-mode D#
    g0[1] = lds_off;                                         // lds_addr
    g0[2] = (unsigned int)ga;                                // global_addr[31:0]
    g0[3] = (unsigned int)((ga >> 32) & 0x1FFFFFFu)          // global_addr[56:32]
          | (2u << 30);                                      // type=2 ("image")
    tdm_v4i z4 = {0, 0, 0, 0};
#if defined(__clang_major__) && (__clang_major__ >= 23)
    tdm_v8i z8 = {0, 0, 0, 0, 0, 0, 0, 0};
    __builtin_amdgcn_tensor_load_to_lds(g0, g1, z4, z4, z8, 0);
#else
    __builtin_amdgcn_tensor_load_to_lds(g0, g1, z4, z4, 0);
#endif
}
#else
#define HAVE_TDM 0
#endif

// ---------------------------------------------------------------------------
// Kernel 1: QKV projection  out = hid @ W^T + b  (bf16 WMMA, f32 accumulate)
// grid = (12, 768, 3), block = 128 (4 waves, each owns one 16x16 N-tile).
// Q,K stored row-major bf16; V stored per-head TRANSPOSED (Vt[b][h][dh][s])
// so the attention P@V B-fragments become contiguous b32 loads; V tail rows
// (s >= 640) are also written straight to the f32 output.
// ---------------------------------------------------------------------------
__global__ __launch_bounds__(128) void qkv_wmma(
    const float* __restrict__ hid,
    const float* __restrict__ Wq, const float* __restrict__ bq,
    const float* __restrict__ Wk, const float* __restrict__ bk,
    const float* __restrict__ Wv, const float* __restrict__ bv,
    unsigned short* __restrict__ Qb, unsigned short* __restrict__ Kb,
    unsigned short* __restrict__ Vt, float* __restrict__ out)
{
    const int lane = threadIdx.x & 31;
    const int wv   = threadIdx.x >> 5;
    const int n0   = (blockIdx.x * 4 + wv) * 16;
    const int m0   = blockIdx.y * 16;
    const int which = blockIdx.z;

    const float* W  = (which == 0) ? Wq : (which == 1) ? Wk : Wv;
    const float* bb = (which == 0) ? bq : (which == 1) ? bk : bv;

    const int hsel = lane >> 4;     // which 16-lane half
    const int mn   = lane & 15;

    const float* arow = hid + (size_t)(m0 + mn) * H_;
    const float* brow = W   + (size_t)(n0 + mn) * H_;

    v8f acc = {0.f, 0.f, 0.f, 0.f, 0.f, 0.f, 0.f, 0.f};
    for (int k0 = 0; k0 < H_; k0 += 32) {
        Frag fa, fb;
#pragma unroll
        for (int j = 0; j < 8; ++j) {
            // A 16x32 bf16 layout: vgpr j holds K pair (ISA 7.12.2)
            const int ka = (j < 4) ? (hsel * 8 + 2 * j) : (16 + hsel * 8 + 2 * (j - 4));
            const float2 av = *(const float2*)(arow + k0 + ka);
            fa.u[j] = packbf2(av.x, av.y);
            // B 32x16 bf16 layout: same K pairs as A
            const int kb = (j < 4) ? (hsel * 8 + 2 * j) : (16 + hsel * 8 + 2 * (j - 4));
            const float2 bv2 = *(const float2*)(brow + k0 + kb);
            fb.u[j] = packbf2(bv2.x, bv2.y);
        }
        acc = __builtin_amdgcn_wmma_f32_16x16x32_bf16(false, fa.v, false, fb.v,
                                                      (short)0, acc, false, false);
    }

    const float bias = bb[n0 + mn];
#pragma unroll
    for (int r = 0; r < 8; ++r) {
        const int row = m0 + r + hsel * 8;       // C/D: vgpr r -> M = r (+8 upper half)
        const int col = n0 + mn;
        const float val = acc[r] + bias;
        if (which == 0) {
            Qb[(size_t)row * H_ + col] = f2bf(val);
        } else if (which == 1) {
            Kb[(size_t)row * H_ + col] = f2bf(val);
        } else {
            const int bidx = row / S_, sloc = row % S_;
            const int hh = col >> 6, dd = col & 63;
            Vt[(((size_t)bidx * NH_ + hh) * DH_ + dd) * S_ + sloc] = f2bf(val);
            if (sloc >= QLEN_)                   // tail rows of output are raw V
                out[(size_t)row * H_ + col] = val;
        }
    }
}

// ---------------------------------------------------------------------------
// Kernel 2: masked attention, flash-style online softmax.
// grid = (10, 12, 16) = (qchunk, head, batch), block = 128 (4 waves x 16 rows).
// DOUBLE-BUFFERED TDM pipeline: wave 0 issues tile i+1's K (32x64) and Vt
// (64x32) D# loads into the alternate LDS buffer, then s_wait_tensorcnt 2
// (in-order completion => tile i done, tile i+1 still in flight), barrier,
// all waves consume tile i from LDS while the DMA engine fills tile i+1.
// Masked scores are 0 (reference multiplies scaled scores by the mask), so
// all 768 keys join the softmax.
// ---------------------------------------------------------------------------
__global__ __launch_bounds__(128) void attn_wmma(
    const unsigned short* __restrict__ Qb,
    const unsigned short* __restrict__ Kb,
    const unsigned short* __restrict__ Vt,
    float* __restrict__ out)
{
    __shared__ __align__(16) unsigned short tK[2][32][64];   // [buf][key][kdim]
    __shared__ __align__(16) unsigned short tV[2][64][32];   // [buf][dh][key]
    __shared__ __align__(16) unsigned short Pld[4][16][32];

    const int lane = threadIdx.x & 31;
    const int wv   = threadIdx.x >> 5;
    const int qc   = blockIdx.x;                 // query chunk 0..9
    const int h    = blockIdx.y;
    const int b    = blockIdx.z;
    const int hsel = lane >> 4;
    const int mn   = lane & 15;

    const int q0 = qc * 64 + wv * 16;
    const size_t baseQ  = ((size_t)b * S_ + q0) * H_ + h * DH_;
    const size_t baseK  = ((size_t)b * S_) * H_ + h * DH_;
    const size_t baseVt = ((size_t)b * NH_ + h) * DH_ * S_;

#if HAVE_TDM
    const tdm_v8i g1K = tdm_g1(64u, 32u, (unsigned int)H_);
    const tdm_v8i g1V = tdm_g1(32u, 64u, (unsigned int)S_);
    auto issue_tile = [&](int ti) {
        const int bs = ti & 1;
        tdm_issue((unsigned int)(size_t)&tK[bs][0][0],
                  Kb + baseK + (size_t)ti * 32 * H_, g1K);
        tdm_issue((unsigned int)(size_t)&tV[bs][0][0],
                  Vt + baseVt + ti * 32, g1V);
    };
    if (wv == 0) issue_tile(0);                  // prologue: tile 0 in flight
#endif

    // Q fragments for K-dim 64 -> two 16x32 A-fragments (from global, once)
    Frag aq[2];
#pragma unroll
    for (int kk = 0; kk < 2; ++kk)
#pragma unroll
        for (int j = 0; j < 8; ++j) {
            const int ka = (j < 4) ? (hsel * 8 + 2 * j) : (16 + hsel * 8 + 2 * (j - 4));
            aq[kk].u[j] = *(const unsigned int*)(Qb + baseQ + (size_t)mn * H_ + kk * 32 + ka);
        }

    const float scale = 0.125f;                  // 1/sqrt(64)
    float m_i[8], l_i[8];
#pragma unroll
    for (int r = 0; r < 8; ++r) { m_i[r] = -3.0e38f; l_i[r] = 0.f; }
    v8f acc[4] = {};

    for (int it = 0; it < NKT_; ++it) {
        const int kb0  = it * 32;
        const int bsel = it & 1;

#if HAVE_TDM
        if (wv == 0) {
            if (it + 1 < NKT_) {
                issue_tile(it + 1);                       // overlap next DMA
                __builtin_amdgcn_s_wait_tensorcnt(2);     // tile `it` complete
            } else {
                __builtin_amdgcn_s_wait_tensorcnt(0);     // drain
            }
        }
#else
        {   // cooperative fallback: 128 threads copy 4KB+4KB with b128 loads
            const int t8 = threadIdx.x * 8;               // 8 bf16 per chunk
            *(uint4*)(&tK[bsel][0][0] + t8)        = *(const uint4*)(Kb + baseK + (size_t)(t8 >> 6) * H_ + (t8 & 63) + (size_t)kb0 * H_);
            *(uint4*)(&tK[bsel][0][0] + 1024 + t8) = *(const uint4*)(Kb + baseK + (size_t)((t8 + 1024) >> 6) * H_ + (t8 & 63) + (size_t)kb0 * H_);
            *(uint4*)(&tV[bsel][0][0] + t8)        = *(const uint4*)(Vt + baseVt + (size_t)(t8 >> 5) * S_ + (t8 & 31) + kb0);
            *(uint4*)(&tV[bsel][0][0] + 1024 + t8) = *(const uint4*)(Vt + baseVt + (size_t)((t8 + 1024) >> 5) * S_ + (t8 & 31) + kb0);
        }
#endif
        __syncthreads();

        // ---- scores: two 16x16 tiles over 32 keys (K frags from LDS) ----
        v8f s0 = {}, s1 = {};
#pragma unroll
        for (int kk = 0; kk < 2; ++kk) {
            Frag fk0, fk1;
#pragma unroll
            for (int j = 0; j < 8; ++j) {
                const int kd = (j < 4) ? (hsel * 8 + 2 * j) : (16 + hsel * 8 + 2 * (j - 4));
                fk0.u[j] = *(const unsigned int*)(&tK[bsel][mn][kk * 32 + kd]);
                fk1.u[j] = *(const unsigned int*)(&tK[bsel][16 + mn][kk * 32 + kd]);
            }
            s0 = __builtin_amdgcn_wmma_f32_16x16x32_bf16(false, aq[kk].v, false, fk0.v,
                                                         (short)0, s0, false, false);
            s1 = __builtin_amdgcn_wmma_f32_16x16x32_bf16(false, aq[kk].v, false, fk1.v,
                                                         (short)0, s1, false, false);
        }

        // scale + mask (key column is per-lane, uniform over rows)
        const int key0 = kb0 + mn, key1 = kb0 + 16 + mn;
        const bool keep0 = (key0 >= QLEN_) || ((key0 >> 6) == qc);
        const bool keep1 = (key1 >= QLEN_) || ((key1 >> 6) == qc);

        // ---- online softmax per row (row r lives in vgpr r of each half) ----
#pragma unroll
        for (int r = 0; r < 8; ++r) {
            const float v0 = keep0 ? s0[r] * scale : 0.f;
            const float v1 = keep1 ? s1[r] * scale : 0.f;
            float t = fmaxf(v0, v1);
            t = fmaxf(t, __shfl_xor(t, 1, 32));
            t = fmaxf(t, __shfl_xor(t, 2, 32));
            t = fmaxf(t, __shfl_xor(t, 4, 32));
            t = fmaxf(t, __shfl_xor(t, 8, 32));
            const float mnew = fmaxf(m_i[r], t);
            const float fr   = __expf(m_i[r] - mnew);
            m_i[r] = mnew;
            const float p0 = __expf(v0 - mnew);
            const float p1 = __expf(v1 - mnew);
            float ps = p0 + p1;
            ps += __shfl_xor(ps, 1, 32);
            ps += __shfl_xor(ps, 2, 32);
            ps += __shfl_xor(ps, 4, 32);
            ps += __shfl_xor(ps, 8, 32);
            l_i[r] = l_i[r] * fr + ps;
            acc[0][r] *= fr; acc[1][r] *= fr; acc[2][r] *= fr; acc[3][r] *= fr;
            // stash P (D-layout) into per-wave LDS for A-layout reload
            const int row = r + hsel * 8;
            Pld[wv][row][mn]      = f2bf(p0);
            Pld[wv][row][16 + mn] = f2bf(p1);
        }
        asm volatile("s_wait_dscnt 0x0" ::: "memory");   // per-wave LDS ordering

        // ---- P (16x32) as A-fragment from LDS ----
        Frag pf;
#pragma unroll
        for (int j = 0; j < 8; ++j) {
            const int ka = (j < 4) ? (hsel * 8 + 2 * j) : (16 + hsel * 8 + 2 * (j - 4));
            pf.u[j] = *(const unsigned int*)(&Pld[wv][mn][ka]);
        }

        // ---- ctx += P @ V : four 16x16 N-tiles over DH=64 (V frags from LDS) ----
#pragma unroll
        for (int t = 0; t < 4; ++t) {
            Frag fv;
#pragma unroll
            for (int j = 0; j < 8; ++j) {
                const int kd = (j < 4) ? (hsel * 8 + 2 * j) : (16 + hsel * 8 + 2 * (j - 4));
                fv.u[j] = *(const unsigned int*)(&tV[bsel][t * 16 + mn][kd]);
            }
            acc[t] = __builtin_amdgcn_wmma_f32_16x16x32_bf16(false, pf.v, false, fv.v,
                                                             (short)0, acc[t], false, false);
        }
        __syncthreads();   // buffer consumed; free for tile it+2's DMA
    }

    // ---- epilogue: normalize and store f32 ----
#pragma unroll
    for (int r = 0; r < 8; ++r) {
        const float inv = 1.0f / l_i[r];
        const int row = q0 + r + hsel * 8;
        const size_t o = ((size_t)b * S_ + row) * H_ + h * DH_;
#pragma unroll
        for (int t = 0; t < 4; ++t)
            out[o + t * 16 + mn] = acc[t][r] * inv;
    }
}

// ---------------------------------------------------------------------------
extern "C" void kernel_launch(void* const* d_in, const int* in_sizes, int n_in,
                              void* d_out, int out_size, void* d_ws, size_t ws_size,
                              hipStream_t stream) {
    const float* hid = (const float*)d_in[0];
    const float* Wq  = (const float*)d_in[1];
    const float* bq  = (const float*)d_in[2];
    const float* Wk  = (const float*)d_in[3];
    const float* bk  = (const float*)d_in[4];
    const float* Wv  = (const float*)d_in[5];
    const float* bv  = (const float*)d_in[6];
    float* out = (float*)d_out;

    const size_t elems = (size_t)B_ * S_ * H_;       // 9.4M
    unsigned short* Qb = (unsigned short*)d_ws;      // bf16 workspaces
    unsigned short* Kb = Qb + elems;
    unsigned short* Vt = Kb + elems;                 // per-head transposed V

    dim3 gA(H_ / 64, (B_ * S_) / 16, 3);             // (12, 768, 3)
    qkv_wmma<<<gA, 128, 0, stream>>>(hid, Wq, bq, Wk, bk, Wv, bv, Qb, Kb, Vt, out);

    dim3 gB(QLEN_ / 64, NH_, B_);                    // (10, 12, 16)
    attn_wmma<<<gB, 128, 0, stream>>>(Qb, Kb, Vt, out);
}
